// DeformableRefine_22445499089568
// MI455X (gfx1250) — matrix operand order repaired and checked by reading, not compiled, not run on hardware
//
#include <hip/hip_runtime.h>
#include <cstdint>
#include <cstddef>

// ---------------------------------------------------------------------------
// DeformableRefine on MI455X (gfx1250): U-Net convs as f16 WMMA implicit GEMM
// (v_wmma_f32_16x16x32_f16). NHWC f16 activations (channels padded to mult.
// of 32), weights repacked to [tap][cout][cinP] f16.
// Conv kernel: block = 16 cout x (16x16 pixel patch); per 32-channel slice the
// (16+halo)^2 patch is staged in LDS once and all 9 taps are served from LDS
// (9x cut in global activation traffic). Wave = 4 independent accumulators
// (4 patch rows) sharing each A fragment.
// ---------------------------------------------------------------------------

typedef __attribute__((ext_vector_type(16))) _Float16 v16h;
typedef __attribute__((ext_vector_type(8)))  _Float16 v8h;
typedef __attribute__((ext_vector_type(8)))  float    v8f;

#define INV_SQRT_BN 0.99999500003749969f   // 1/sqrt(1 + 1e-5)

// ---------------- weight repack: OIHW f32 -> [tap][cout][cinP] f16 ----------
__global__ void k_pack_w(const float* __restrict__ w, _Float16* __restrict__ wp,
                         int Cin, int cpsh, int cosh_, int K2, int total) {
  int i = blockIdx.x * blockDim.x + threadIdx.x;
  if (i >= total) return;
  int ci = i & ((1 << cpsh) - 1);
  int t  = i >> cpsh;
  int co = t & ((1 << cosh_) - 1);
  int tap = t >> cosh_;
  float v = (ci < Cin) ? w[((size_t)co * Cin + ci) * K2 + tap] : 0.0f;
  wp[i] = (_Float16)v;
}

// ---------------- img [4,3,512,512] f32 -> NHWC f16 padded to C=32 ----------
__global__ void k_img_nhwc(const float* __restrict__ img, _Float16* __restrict__ out) {
  int i = blockIdx.x * blockDim.x + threadIdx.x;   // total = 4*512*512*32
  int c = i & 31;
  int p = i >> 5;
  int x = p & 511;
  int y = (p >> 9) & 511;
  int b = p >> 18;
  float v = (c < 3) ? img[(((size_t)b * 3 + c) << 18) + (y << 9) + x] : 0.0f;
  out[i] = (_Float16)v;
}

// ---------------- WMMA conv (KS=1 or 3, SAME), NHWC f16, LDS-tiled ----------
// x: [4,H,W,CIN]  wp: [KS*KS, Cout, CIN] f16  y: [4,H,W,Cout]
// H == W == 1<<wsh (power of two), cosh = log2(Cout).
// Block = 128 threads (4 waves): 16 cout x 16x16 patch. Wave w owns patch
// rows 4w..4w+3 (one 16-pixel accumulator per row). Per 32-ch slice, the
// (16+2*HALO)^2 halo patch is staged into LDS (pixel stride 40 halves = 80B,
// 16B-aligned v16h, bank-spread), then the KS*KS taps stream from LDS.
template <int CIN, int KS>
__global__ __launch_bounds__(128)
void k_conv_wmma(const _Float16* __restrict__ x, const _Float16* __restrict__ wp,
                 const float* __restrict__ cbias, const float* __restrict__ gamma,
                 const float* __restrict__ beta, _Float16* __restrict__ y,
                 int wsh, int cosh_) {
  constexpr int K2   = KS * KS;
  constexpr int HALO = KS / 2;
  constexpr int TS   = 16 + 2 * HALO;      // staged patch side
  constexpr int PST  = 40;                 // halves per staged pixel (32 + pad)
  constexpr int CSH  = (CIN == 32) ? 5 : (CIN == 64) ? 6 : (CIN == 128) ? 7 : 8;

  __shared__ _Float16 tile[TS * TS * PST];

  const int tid  = threadIdx.x;
  const int lane = tid & 31;
  const int wave = tid >> 5;
  const int grp  = lane >> 4;
  const int lid  = lane & 15;
  const int W    = 1 << wsh;
  const int b    = blockIdx.z;
  const int coBase = blockIdx.y << 4;
  const int wt = wsh - 4;                       // log2(tiles per row)
  const int x0 = (blockIdx.x & ((1 << wt) - 1)) << 4;
  const int y0 = (blockIdx.x >> wt) << 4;
  const size_t bBase = (size_t)b << (2 * wsh);
  const int coA = coBase + lid;                 // this lane's A (weight) row

  v8f acc0 = {}, acc1 = {}, acc2 = {}, acc3 = {};

#pragma unroll
  for (int ci0 = 0; ci0 < CIN; ci0 += 32) {
    __syncthreads();
    // ---- stage (TS x TS) x 32ch patch into LDS (zero-filled halo) ----
    for (int i = tid; i < TS * TS; i += 128) {
      const int py = i / TS;
      const int px = i % TS;
      const int gy = y0 + py - HALO;
      const int gx = x0 + px - HALO;
      v16h lo = {}, hi = {};
      if (gy >= 0 && gy < W && gx >= 0 && gx < W) {
        const _Float16* src = x + ((bBase + ((size_t)gy << wsh) + gx) << CSH) + ci0;
        lo = *(const v16h*)(src);
        hi = *(const v16h*)(src + 16);
      }
      *(v16h*)&tile[i * PST]      = lo;
      *(v16h*)&tile[i * PST + 16] = hi;
    }
    __syncthreads();

    // ---- KS*KS taps from LDS ----
    for (int tap = 0; tap < K2; ++tap) {
      const int ky = tap / KS;                  // 0..KS-1 (halo-relative)
      const int kx = tap % KS;
      const _Float16* wrow = wp + ((((size_t)tap << cosh_) + coA) << CSH) + ci0;
      // A fragment (16x32 f16), lane row M = lid.
      // halves 0..15 -> K = grp*16 + 0..15 (contiguous, matches B)
      v16h a = *(const v16h*)(wrow + grp * 16);
      // B fragments (32x16 f16): lane col N = lid -> patch col lid+kx.
      // halves 0..15 -> K = grp*16 + 0..15 (contiguous channels)
      const int col = lid + kx;
      const int r0  = (wave << 2) + ky;
      const _Float16* t0 = &tile[((r0 + 0) * TS + col) * PST + grp * 16];
      const _Float16* t1 = &tile[((r0 + 1) * TS + col) * PST + grp * 16];
      const _Float16* t2 = &tile[((r0 + 2) * TS + col) * PST + grp * 16];
      const _Float16* t3 = &tile[((r0 + 3) * TS + col) * PST + grp * 16];
      v16h b0 = *(const v16h*)t0;
      v16h b1 = *(const v16h*)t1;
      v16h b2 = *(const v16h*)t2;
      v16h b3 = *(const v16h*)t3;
      acc0 = __builtin_amdgcn_wmma_f32_16x16x32_f16(false, a, false, b0, (short)0, acc0, false, false);
      acc1 = __builtin_amdgcn_wmma_f32_16x16x32_f16(false, a, false, b1, (short)0, acc1, false, false);
      acc2 = __builtin_amdgcn_wmma_f32_16x16x32_f16(false, a, false, b2, (short)0, acc2, false, false);
      acc3 = __builtin_amdgcn_wmma_f32_16x16x32_f16(false, a, false, b3, (short)0, acc3, false, false);
    }
  }

  // ---- epilogue: relu(acc*scale + shift), f16 NHWC store ----
  // D layout: vgpr r, lane -> M = r + grp*8, N = lid.
  const int coL = coBase + (grp << 3);
  float sc[8], sh[8];
#pragma unroll
  for (int r = 0; r < 8; ++r) {
    const int co = coL + r;
    const float s = gamma ? gamma[co] * INV_SQRT_BN : 1.0f;
    sc[r] = s;
    sh[r] = (cbias ? cbias[co] : 0.0f) * s + (gamma ? beta[co] : 0.0f);
  }
  v8f accs[4] = {acc0, acc1, acc2, acc3};
#pragma unroll
  for (int t = 0; t < 4; ++t) {
    v8h outv;
#pragma unroll
    for (int r = 0; r < 8; ++r) {
      float v = fmaf(accs[t][r], sc[r], sh[r]);
      v = v > 0.0f ? v : 0.0f;
      outv[r] = (_Float16)v;
    }
    const size_t p = bBase + (((size_t)(y0 + (wave << 2) + t)) << wsh) + x0 + lid;
    *(v8h*)(y + (p << cosh_) + coL) = outv;
  }
}

// ---------------- 2x2 max pool, NHWC f16 (square maps) ----------------------
__global__ void k_maxpool2(const _Float16* __restrict__ x, _Float16* __restrict__ y,
                           int csh, int wosh, int total) {
  int i = blockIdx.x * blockDim.x + threadIdx.x;
  if (i >= total) return;
  const int C  = 1 << csh;
  int c  = i & (C - 1);
  int p  = i >> csh;
  int xo = p & ((1 << wosh) - 1);
  int yo = (p >> wosh) & ((1 << wosh) - 1);
  int b  = p >> (2 * wosh);
  const int iwsh = wosh + 1;
  size_t base = ((((size_t)b << (2 * iwsh)) + ((size_t)(yo << 1) << iwsh) + (xo << 1)) << csh) + c;
  float v0 = (float)x[base];
  float v1 = (float)x[base + C];
  float v2 = (float)x[base + ((size_t)1 << (iwsh + csh))];
  float v3 = (float)x[base + ((size_t)1 << (iwsh + csh)) + C];
  y[i] = (_Float16)fmaxf(fmaxf(v0, v1), fmaxf(v2, v3));
}

// ---------------- nearest 2x upsample, NHWC f16 (square maps) ---------------
__global__ void k_up2(const _Float16* __restrict__ x, _Float16* __restrict__ y,
                      int csh, int wosh, int total) {
  int i = blockIdx.x * blockDim.x + threadIdx.x;
  if (i >= total) return;
  int c  = i & ((1 << csh) - 1);
  int p  = i >> csh;
  int xo = p & ((1 << wosh) - 1);
  int yo = (p >> wosh) & ((1 << wosh) - 1);
  int b  = p >> (2 * wosh);
  const int iwsh = wosh - 1;
  y[i] = x[((((size_t)b << (2 * iwsh)) + ((size_t)(yo >> 1) << iwsh) + (xo >> 1)) << csh) + c];
}

// ---------------- channel concat 128|128 -> 256, NHWC f16 -------------------
__global__ void k_concat128(const _Float16* __restrict__ A, const _Float16* __restrict__ Bx,
                            _Float16* __restrict__ y, int total) {
  int i = blockIdx.x * blockDim.x + threadIdx.x;
  if (i >= total) return;
  int c = i & 255;
  int p = i >> 8;
  y[i] = (c < 128) ? A[((size_t)p << 7) + c] : Bx[((size_t)p << 7) + (c - 128)];
}

// ---------------- 1x1 heads: offset(18ch) + sigmoid mask(9ch), 128x128 ------
__global__ void k_heads(const _Float16* __restrict__ feat,   // [4,16384,32] NHWC
                        const float* __restrict__ pw, const float* __restrict__ pb,
                        const float* __restrict__ mw, const float* __restrict__ mb,
                        float* __restrict__ off_out,         // [4,18,16384]
                        float* __restrict__ m_out) {         // [4,9,16384]
  constexpr int HW = 16384;
  int i = blockIdx.x * blockDim.x + threadIdx.x;
  if (i >= 4 * HW) return;
  int b = i >> 14;
  int p = i & (HW - 1);
  float f[32];
#pragma unroll
  for (int c = 0; c < 32; ++c) f[c] = (float)feat[((size_t)i << 5) + c];
  for (int n = 0; n < 18; ++n) {
    float s = pb[n];
#pragma unroll
    for (int c = 0; c < 32; ++c) s += pw[n * 32 + c] * f[c];
    off_out[((size_t)(b * 18 + n) << 14) + p] = s;
  }
  for (int n = 0; n < 9; ++n) {
    float s = mb[n];
#pragma unroll
    for (int c = 0; c < 32; ++c) s += mw[n * 32 + c] * f[c];
    m_out[((size_t)(b * 9 + n) << 14) + p] = 1.0f / (1.0f + __expf(-s));
  }
}

// ---------------- deformable bilinear sample + weighted mean ----------------
__global__ void k_sample(const float* __restrict__ depth, const float* __restrict__ off,
                         const float* __restrict__ m, float* __restrict__ out) {
  constexpr int HW = 16384, H = 512, W = 512;
  int i = blockIdx.x * blockDim.x + threadIdx.x;
  if (i >= 4 * HW) return;
  int b  = i >> 14;
  int p  = i & (HW - 1);
  int oy = p >> 7;
  int ox = p & 127;
  const float* dimg = depth + ((size_t)b << 18);
  float num = 0.0f, den = 0.0f;
  for (int n = 0; n < 9; ++n) {
    float offy = off[((size_t)(b * 18 + n) << 14) + p];
    float offx = off[((size_t)(b * 18 + 9 + n) << 14) + p];
    float py = (float)(oy + 1) + offy;
    float px = (float)(ox + 1) + offx;
    float gy = fminf(fmaxf(py, 0.0f), 511.0f) * (2.0f / 511.0f) - 1.0f;
    float gx = fminf(fmaxf(px, 0.0f), 511.0f) * (2.0f / 511.0f) - 1.0f;
    float iy = ((gy + 1.0f) * (float)H - 1.0f) * 0.5f;
    float ix = ((gx + 1.0f) * (float)W - 1.0f) * 0.5f;
    float y0f = floorf(iy), x0f = floorf(ix);
    float wx1 = ix - x0f, wx0 = 1.0f - wx1;
    float wy1 = iy - y0f, wy0 = 1.0f - wy1;
    int y0 = (int)y0f, x0 = (int)x0f;
    float val = 0.0f;
#pragma unroll
    for (int dy = 0; dy < 2; ++dy) {
#pragma unroll
      for (int dx = 0; dx < 2; ++dx) {
        int yc = y0 + dy, xc = x0 + dx;
        bool valid = (yc >= 0) && (yc < H) && (xc >= 0) && (xc < W);
        int yi = yc < 0 ? 0 : (yc > H - 1 ? H - 1 : yc);
        int xi = xc < 0 ? 0 : (xc > W - 1 ? W - 1 : xc);
        float g = valid ? dimg[((size_t)yi << 9) + xi] : 0.0f;
        val += (dy ? wy1 : wy0) * (dx ? wx1 : wx0) * g;
      }
    }
    float mm = m[((size_t)(b * 9 + n) << 14) + p];
    num += mm * val;
    den += mm;
  }
  out[i] = num / (den + 1e-8f);
}

// ---------------------------------------------------------------------------
// Host orchestration.
// Input flattening assumption (setup_inputs dict insertion order):
//  0 img[4,3,512,512] 1 depth[4,512,512]
//  2..7 c1 | 8..13 c2 | 14..19 c3 | 20..25 c4 | 26..31 c5   (w1 g1 b1 w2 g2 b2)
//  32..35 u5w u5b u5g u5bb | 36..41 uc5 | 42..45 u4w.. | 46..51 uc4
//  52 fw 53 fb | 54 pw 55 pb | 56 mw 57 mb
// ---------------------------------------------------------------------------
extern "C" void kernel_launch(void* const* d_in, const int* in_sizes, int n_in,
                              void* d_out, int out_size, void* d_ws, size_t ws_size,
                              hipStream_t stream) {
  (void)in_sizes; (void)n_in; (void)out_size; (void)ws_size;

  const float* img   = (const float*)d_in[0];
  const float* depth = (const float*)d_in[1];

  // Workspace layout.
  const size_t S_BIG = (size_t)4 * 512 * 512 * 32 * 2;   // 67,108,864 B
  const size_t S3_SZ = (size_t)4 * 128 * 128 * 128 * 2;  // 16,777,216 B
  const size_t S4_SZ = (size_t)4 * 64 * 64 * 128 * 2;    //  4,194,304 B
  const size_t SM_SZ = (size_t)4 * 9 * 128 * 128 * 4;    //  2,359,296 B
  char* wsb = (char*)d_ws;
  _Float16* S0 = (_Float16*)(wsb);
  _Float16* S1 = (_Float16*)(wsb + S_BIG);
  _Float16* S2 = (_Float16*)(wsb + 2 * S_BIG);
  _Float16* S3 = (_Float16*)(wsb + 3 * S_BIG);
  _Float16* S4 = (_Float16*)(wsb + 3 * S_BIG + S3_SZ);
  float*    SM = (float*)   (wsb + 3 * S_BIG + S3_SZ + S4_SZ);
  _Float16* WP = (_Float16*)(wsb + 3 * S_BIG + S3_SZ + S4_SZ + SM_SZ); // <=1MB

  auto cdiv = [](int a, int b) { return (a + b - 1) / b; };
  auto lg2  = [](int v) { int s = 0; while ((1 << s) < v) ++s; return s; };

  auto conv = [&](const _Float16* xin, int widx, const float* cb, int gidx, int bidx,
                  _Float16* yout, int wsh, int Cin, int CinP, int Cout, int ks) {
    const float* w  = (const float*)d_in[widx];
    const float* g  = (gidx >= 0) ? (const float*)d_in[gidx] : nullptr;
    const float* be = (bidx >= 0) ? (const float*)d_in[bidx] : nullptr;
    const int cosh_ = lg2(Cout);
    const int tw = ks * ks * Cout * CinP;
    k_pack_w<<<cdiv(tw, 256), 256, 0, stream>>>(w, WP, Cin, lg2(CinP), cosh_,
                                                ks * ks, tw);
    dim3 grid(1 << (2 * wsh - 8), Cout >> 4, 4);   // (W/16)^2 tiles
    if (ks == 3) {
      switch (CinP) {
        case 32:
          k_conv_wmma<32, 3><<<grid, 128, 0, stream>>>(xin, WP, cb, g, be, yout, wsh, cosh_);
          break;
        case 64:
          k_conv_wmma<64, 3><<<grid, 128, 0, stream>>>(xin, WP, cb, g, be, yout, wsh, cosh_);
          break;
        case 128:
          k_conv_wmma<128, 3><<<grid, 128, 0, stream>>>(xin, WP, cb, g, be, yout, wsh, cosh_);
          break;
        default:
          k_conv_wmma<256, 3><<<grid, 128, 0, stream>>>(xin, WP, cb, g, be, yout, wsh, cosh_);
          break;
      }
    } else {
      k_conv_wmma<64, 1><<<grid, 128, 0, stream>>>(xin, WP, cb, g, be, yout, wsh, cosh_);
    }
  };
  auto pool = [&](const _Float16* xin, _Float16* yout, int C, int Wo) {
    int t = 4 * Wo * Wo * C;
    k_maxpool2<<<cdiv(t, 256), 256, 0, stream>>>(xin, yout, lg2(C), lg2(Wo), t);
  };
  auto up2 = [&](const _Float16* xin, _Float16* yout, int C, int Wo) {
    int t = 4 * Wo * Wo * C;
    k_up2<<<cdiv(t, 256), 256, 0, stream>>>(xin, yout, lg2(C), lg2(Wo), t);
  };
  auto cat = [&](const _Float16* A, const _Float16* Bx, _Float16* yout, int Wc) {
    int t = 4 * Wc * Wc * 256;
    k_concat128<<<cdiv(t, 256), 256, 0, stream>>>(A, Bx, yout, t);
  };

  // ---- encoder ----
  {
    int t = 4 * 512 * 512 * 32;
    k_img_nhwc<<<cdiv(t, 256), 256, 0, stream>>>(img, S0);
  }
  conv(S0, 2, nullptr, 3, 4, S1, 9, 3, 32, 32, 3);        // c1a  (512^2)
  conv(S1, 5, nullptr, 6, 7, S2, 9, 32, 32, 32, 3);       // x1
  pool(S2, S1, 32, 256);                                  // p1
  conv(S1, 8, nullptr, 9, 10, S2, 8, 32, 32, 64, 3);      // c2a  (256^2)
  conv(S2, 11, nullptr, 12, 13, S1, 8, 64, 64, 64, 3);    // x2
  pool(S1, S2, 64, 128);                                  // p2
  conv(S2, 14, nullptr, 15, 16, S1, 7, 64, 64, 128, 3);   // c3a  (128^2)
  conv(S1, 17, nullptr, 18, 19, S3, 7, 128, 128, 128, 3); // x3 (skip)
  pool(S3, S1, 128, 64);                                  // p3
  conv(S1, 20, nullptr, 21, 22, S2, 6, 128, 128, 128, 3); // c4a  (64^2)
  conv(S2, 23, nullptr, 24, 25, S4, 6, 128, 128, 128, 3); // x4 (skip)
  pool(S4, S1, 128, 32);                                  // p4
  conv(S1, 26, nullptr, 27, 28, S2, 5, 128, 128, 128, 3); // c5a  (32^2)
  conv(S2, 29, nullptr, 30, 31, S1, 5, 128, 128, 128, 3); // x5

  // ---- decoder level 5 ----
  up2(S1, S2, 128, 64);                                   // up5
  conv(S2, 32, (const float*)d_in[33], 34, 35, S1, 6, 128, 128, 128, 3); // d5a
  cat(S4, S1, S2, 64);                                    // cat5 = x4 | d5a
  conv(S2, 36, nullptr, 37, 38, S1, 6, 256, 256, 128, 3); // uc5a
  conv(S1, 39, nullptr, 40, 41, S2, 6, 128, 128, 128, 3); // d5

  // ---- decoder level 4 ----
  up2(S2, S1, 128, 128);                                  // up4
  conv(S1, 42, (const float*)d_in[43], 44, 45, S2, 7, 128, 128, 128, 3); // d4a
  cat(S3, S2, S1, 128);                                   // cat4 = x3 | d4a
  conv(S1, 46, nullptr, 47, 48, S2, 7, 256, 256, 64, 3);  // uc4a
  conv(S2, 49, nullptr, 50, 51, S1, 7, 64, 64, 64, 3);    // d4

  // ---- final 1x1 feature conv: relu(conv + fb), no BN ----
  conv(S1, 52, (const float*)d_in[53], -1, -1, S2, 7, 64, 64, 32, 1);   // feature

  // ---- heads + deformable sampling ----
  const int HW = 16384;
  float* off_out = (float*)d_out + 4 * HW;   // output[1]: offsets [4,18,128,128]
  float* dep_out = (float*)d_out;            // output[0]: depth   [4,1,128,128]
  k_heads<<<cdiv(4 * HW, 128), 128, 0, stream>>>(
      S2, (const float*)d_in[54], (const float*)d_in[55],
      (const float*)d_in[56], (const float*)d_in[57], off_out, SM);
  k_sample<<<cdiv(4 * HW, 128), 128, 0, stream>>>(depth, off_out, SM, dep_out);
}
